// SolveRegistration_56307021251020
// MI455X (gfx1250) — compile-verified, matrix operand order checked by reading, not run on hardware
//
#include <hip/hip_runtime.h>
#include <math.h>

// ---------------- types ----------------
typedef __attribute__((ext_vector_type(16))) _Float16 v16h;
typedef __attribute__((ext_vector_type(8)))  float    v8f;

struct CloudDesc {
  int   in_off;    // point offset into source cloud array
  int   out_off;   // feature offset into output feature base
  float A[9];      // 3x3 affine (row-major)
  float t[3];      // translation (mean subtraction baked in)
  int   pad[2];
}; // 64 bytes

// ---------------- constants ----------------
#define BATCH   8
#define NPTS    2048
#define KFEAT   1024
#define H1DIM   64
#define H2DIM   128
#define ROWS_PER_BLOCK 64
#define ENC_THREADS    128   // 4 waves of 32

// workspace layout (float units)
#define OFF_MEANS   0                          // p0_m[8][3] then p1_m[8][3] : 48
#define OFF_TRANSF  48                         // 6 * 12 : 72
#define OFF_G       120                        // 8 * 12 : 96
#define OFF_FEATB   216                        // f0 (8192) + fj (49152) contiguous
#define OFF_F1      (OFF_FEATB + 57344)        // 8192
#define OFF_PINV    (OFF_F1 + 8192)            // 49152
#define OFF_ESTG    (OFF_PINV + 49152)         // 96
#define OFF_INVIGT  (OFF_ESTG + 96)            // 96
#define OFF_DESCB   (OFF_INVIGT + 96)          // 56 descs * 16 floats : 896
#define OFF_DESC1   (OFF_DESCB + 896)          // 8 descs * 16 floats : 128
#define OFF_W2S     (OFF_DESC1 + 128)          // 8192 halfs = 4096 floats (32B aligned)
#define OFF_W3S     (OFF_W2S + 4096)           // 131072 halfs = 65536 floats

// ---------------- helpers ----------------
__device__ __forceinline__ void atomicMaxFloat(float* addr, float val) {
  // works with addr initialized to -inf bit pattern 0xFF800000
  if (!(val < 0.0f)) atomicMax((int*)addr, __float_as_int(val));
  else               atomicMin((unsigned int*)addr, __float_as_uint(val));
}

__device__ __forceinline__ void se3_exp_dev(const float x[6], float R[9], float p[3]) {
  float wx = x[0], wy = x[1], wz = x[2];
  float vx = x[3], vy = x[4], vz = x[5];
  float t2 = wx*wx + wy*wy + wz*wz;
  float t  = sqrtf(t2);
  bool small = t < 1e-6f;
  float ts = small ? 1.0f : t;
  float A  = small ? (1.0f - t2 * (1.0f/6.0f))   : (sinf(ts)/ts);
  float Bc = small ? (0.5f - t2 * (1.0f/24.0f))  : ((1.0f - cosf(ts))/(ts*ts));
  float Cc = small ? ((1.0f/6.0f) - t2*(1.0f/120.0f)) : ((ts - sinf(ts))/(ts*ts*ts));
  float W[9]  = {0.f, -wz,  wy,   wz, 0.f, -wx,  -wy,  wx, 0.f};
  float W2[9];
  #pragma unroll
  for (int i = 0; i < 3; ++i)
    #pragma unroll
    for (int j = 0; j < 3; ++j) {
      float s = 0.f;
      #pragma unroll
      for (int k = 0; k < 3; ++k) s += W[i*3+k]*W[k*3+j];
      W2[i*3+j] = s;
    }
  float V[9];
  #pragma unroll
  for (int i = 0; i < 9; ++i) {
    float id = (i % 4 == 0) ? 1.0f : 0.0f;
    R[i] = id + A*W[i]  + Bc*W2[i];
    V[i] = id + Bc*W[i] + Cc*W2[i];
  }
  p[0] = V[0]*vx + V[1]*vy + V[2]*vz;
  p[1] = V[3]*vx + V[4]*vy + V[5]*vz;
  p[2] = V[6]*vx + V[7]*vy + V[8]*vz;
}

__device__ __forceinline__ void invert6(const float* Ain, float* out) {
  float M[6][12];
  #pragma unroll
  for (int i = 0; i < 6; ++i)
    for (int j = 0; j < 6; ++j) { M[i][j] = Ain[i*6+j]; M[i][6+j] = (i == j) ? 1.f : 0.f; }
  for (int col = 0; col < 6; ++col) {
    int piv = col; float mx = fabsf(M[col][col]);
    for (int r = col+1; r < 6; ++r) { float a = fabsf(M[r][col]); if (a > mx) { mx = a; piv = r; } }
    if (piv != col)
      for (int j = 0; j < 12; ++j) { float tmp = M[col][j]; M[col][j] = M[piv][j]; M[piv][j] = tmp; }
    float inv = 1.0f / M[col][col];
    for (int j = 0; j < 12; ++j) M[col][j] *= inv;
    for (int r = 0; r < 6; ++r) if (r != col) {
      float f = M[r][col];
      for (int j = 0; j < 12; ++j) M[r][j] -= f * M[col][j];
    }
  }
  for (int i = 0; i < 6; ++i)
    for (int j = 0; j < 6; ++j) out[i*6+j] = M[i][6+j];
}

// WMMA A-fragment gather per CDNA5 16-bit 16x32 layout (wave32):
// lanes 0-15 row=lane hold k0..k0+7 (elems 0-7) and k0+16..k0+23 (elems 8-15);
// lanes 16-31 row=lane-16 hold k0+8.. and k0+24..  -> two contiguous 16B LDS loads.
__device__ __forceinline__ v16h load_A_frag(const _Float16* sm, int rowBase, int stride, int k0, int lane) {
  int row = rowBase + (lane & 15);
  int kh  = k0 + 8 * (lane >> 4);
  const _Float16* p0 = sm + row * stride + kh;
  v16h a;
  #pragma unroll
  for (int e = 0; e < 8; ++e) { a[e] = p0[e]; a[e+8] = p0[e+16]; }
  return a;
}

// ---------------- kernels ----------------

// Pre-swizzle W2 (64x128) and W3 (128x1024) into WMMA B-fragment order:
// [frag][lane][16 halfs] with frag = nt*KT + kt; each lane's fragment is one
// contiguous 32B chunk -> 2x global_load_b128 per fragment in the GEMM loops.
__global__ void wswz_kernel(const float* __restrict__ W2, const float* __restrict__ W3,
                            _Float16* __restrict__ w2s, _Float16* __restrict__ w3s) {
  int idx = blockIdx.x * 256 + threadIdx.x;     // grid covers 8192 + 131072
  if (idx < 8192) {
    int frag = idx >> 9;                        // 16 frags (nt*2 + kt)
    int lane = (idx >> 4) & 31;
    int e    = idx & 15;
    int nt = frag >> 1, kt = frag & 1;
    int n = nt * 16 + (lane & 15);
    int k = kt * 32 + 8 * (lane >> 4) + ((e < 8) ? e : e + 8);
    w2s[idx] = (_Float16)W2[k * H2DIM + n];
  }
  int j = idx - 8192;
  if (j >= 0 && j < 131072) {
    int frag = j >> 9;                          // 256 frags (nt*4 + kt)
    int lane = (j >> 4) & 31;
    int e    = j & 15;
    int nt = frag >> 2, kt = frag & 3;
    int n = nt * 16 + (lane & 15);
    int k = kt * 32 + 8 * (lane >> 4) + ((e < 8) ? e : e + 8);
    w3s[j] = (_Float16)W3[k * KFEAT + n];
  }
}

__global__ void means_kernel(const float* __restrict__ p_tgt, const float* __restrict__ p_src,
                             float* __restrict__ means) {
  int which = blockIdx.x >> 3;                  // 0: p_tgt -> p0_m, 1: p_src -> p1_m
  int b = blockIdx.x & 7;
  int tid = threadIdx.x;
  const float* src = which ? p_src : p_tgt;
  float sx = 0.f, sy = 0.f, sz = 0.f;
  for (int n = tid; n < NPTS; n += 256) {
    int g = (b * NPTS + n) * 3;
    sx += src[g]; sy += src[g+1]; sz += src[g+2];
  }
  __shared__ float red[256 * 3];
  red[tid] = sx; red[256 + tid] = sy; red[512 + tid] = sz;
  __syncthreads();
  for (int s = 128; s > 0; s >>= 1) {
    if (tid < s) {
      red[tid]       += red[tid + s];
      red[256 + tid] += red[256 + tid + s];
      red[512 + tid] += red[512 + tid + s];
    }
    __syncthreads();
  }
  if (tid == 0) {
    float* m = means + which * 24 + b * 3;
    m[0] = red[0]   * (1.0f / NPTS);
    m[1] = red[256] * (1.0f / NPTS);
    m[2] = red[512] * (1.0f / NPTS);
  }
}

__global__ void init_kernel(float* __restrict__ featB, float* __restrict__ loss_out) {
  int i = blockIdx.x * 256 + threadIdx.x;       // grid covers 57344 floats (f0 + fj)
  ((unsigned int*)featB)[i] = 0xFF800000u;      // -inf
  if (i == 0) *loss_out = 0.0f;
}

__global__ void prepB_kernel(const float* __restrict__ dt, const float* __restrict__ means,
                             float* __restrict__ transf, float* __restrict__ g,
                             CloudDesc* __restrict__ descs) {
  int tid = threadIdx.x;
  if (tid < 6) {
    float x[6] = {0.f,0.f,0.f,0.f,0.f,0.f};
    x[tid] = -dt[tid];
    float R[9], p[3];
    se3_exp_dev(x, R, p);
    float* o = transf + tid * 12;
    #pragma unroll
    for (int i = 0; i < 9; ++i) o[i] = R[i];
    o[9] = p[0]; o[10] = p[1]; o[11] = p[2];
  }
  if (tid >= 8 && tid < 16) {                   // g = identity per batch
    float* gb = g + (tid - 8) * 12;
    #pragma unroll
    for (int i = 0; i < 9; ++i) gb[i] = (i % 4 == 0) ? 1.f : 0.f;
    gb[9] = gb[10] = gb[11] = 0.f;
  }
  __syncthreads();
  if (tid < 56) {
    CloudDesc d;
    d.pad[0] = d.pad[1] = 0;
    if (tid < 8) {                              // f0 clouds: q0 = p_tgt - p0_m
      int b = tid;
      const float* m = means + b * 3;
      d.in_off  = b * NPTS;
      d.out_off = b * KFEAT;
      #pragma unroll
      for (int i = 0; i < 9; ++i) d.A[i] = (i % 4 == 0) ? 1.f : 0.f;
      d.t[0] = -m[0]; d.t[1] = -m[1]; d.t[2] = -m[2];
    } else {                                    // fj clouds: R_j (raw - p0_m) + t_j
      int c = tid - 8, b = c / 6, j = c % 6;
      const float* T = transf + j * 12;
      const float* m = means + b * 3;
      d.in_off  = b * NPTS;
      d.out_off = BATCH * KFEAT + (b * 6 + j) * KFEAT;
      #pragma unroll
      for (int i = 0; i < 9; ++i) d.A[i] = T[i];
      #pragma unroll
      for (int i = 0; i < 3; ++i)
        d.t[i] = T[9 + i] - (T[i*3]*m[0] + T[i*3+1]*m[1] + T[i*3+2]*m[2]);
    }
    descs[tid] = d;
  }
}

__global__ void iter_setup_kernel(const float* __restrict__ g, const float* __restrict__ means,
                                  CloudDesc* __restrict__ descs, float* __restrict__ f1) {
  int tid = threadIdx.x;
  for (int i = tid; i < BATCH * KFEAT; i += 256)
    ((unsigned int*)f1)[i] = 0xFF800000u;       // -inf
  if (tid < 8) {
    int b = tid;
    const float* gb = g + b * 12;
    const float* m  = means + 24 + b * 3;       // p1_m
    CloudDesc d;
    d.pad[0] = d.pad[1] = 0;
    d.in_off  = b * NPTS;
    d.out_off = b * KFEAT;
    #pragma unroll
    for (int i = 0; i < 9; ++i) d.A[i] = gb[i];
    #pragma unroll
    for (int i = 0; i < 3; ++i)
      d.t[i] = gb[9 + i] - (gb[i*3]*m[0] + gb[i*3+1]*m[1] + gb[i*3+2]*m[2]);
    descs[tid] = d;
  }
}

// Fused MLP encode + maxpool. One block = 64 points, 4 waves; each wave owns 16 rows.
// Layer1 (3->64) in VALU; Layer2 (64->128) and Layer3 (128->1024) on
// v_wmma_f32_16x16x32_f16. A-fragments are hoisted into registers (invariant
// across n-tiles); B-fragments are pre-swizzled in global memory so each lane
// issues two coalesced global_load_b128 per fragment (L2-resident weights).
// Maxpool fused into the layer-3 epilogue via int atomicMax/uint atomicMin on float bits; b3
// cancels algebraically so it is skipped.
__global__ void __launch_bounds__(ENC_THREADS)
encode_kernel(const float* __restrict__ pts, const CloudDesc* __restrict__ descs,
              float* __restrict__ outbase,
              const float* __restrict__ W1, const float* __restrict__ b1,
              const float* __restrict__ b2,
              const _Float16* __restrict__ w2s, const _Float16* __restrict__ w3s) {
  __shared__ _Float16 sH1[ROWS_PER_BLOCK * H1DIM];   // 8 KB
  __shared__ _Float16 sH2[ROWS_PER_BLOCK * H2DIM];   // 16 KB
  __shared__ float    sTP[ROWS_PER_BLOCK * 3];       // transformed points

  const int cloud   = blockIdx.x >> 5;          // 32 row-tiles per cloud (2048/64)
  const int rowtile = blockIdx.x & 31;
  const CloudDesc d = descs[cloud];

  const int tid  = threadIdx.x;
  const int lane = tid & 31;
  const int rlo  = (tid >> 5) * 16;             // local row base for this wave

  // --- transform points ---
  if (tid < ROWS_PER_BLOCK) {
    int gidx = (d.in_off + rowtile * ROWS_PER_BLOCK + tid) * 3;
    float x = pts[gidx], y = pts[gidx+1], z = pts[gidx+2];
    sTP[tid*3+0] = d.A[0]*x + d.A[1]*y + d.A[2]*z + d.t[0];
    sTP[tid*3+1] = d.A[3]*x + d.A[4]*y + d.A[5]*z + d.t[1];
    sTP[tid*3+2] = d.A[6]*x + d.A[7]*y + d.A[8]*z + d.t[2];
  }
  __syncthreads();

  // --- layer 1: 3 -> 64 (VALU), paired f16 stores ---
  for (int i = tid; i < ROWS_PER_BLOCK * H1DIM / 2; i += ENC_THREADS) {
    int row = i >> 5, cp = i & 31;
    int col = cp * 2;
    float px = sTP[row*3+0], py = sTP[row*3+1], pz = sTP[row*3+2];
    float a0 = b1[col]   + px * W1[col]   + py * W1[H1DIM + col]   + pz * W1[2*H1DIM + col];
    float a1 = b1[col+1] + px * W1[col+1] + py * W1[H1DIM + col+1] + pz * W1[2*H1DIM + col+1];
    union { _Float16 h[2]; unsigned u; } pk;
    pk.h[0] = (_Float16)fmaxf(a0, 0.0f);
    pk.h[1] = (_Float16)fmaxf(a1, 0.0f);
    ((unsigned*)sH1)[row * 32 + cp] = pk.u;
  }
  __syncthreads();

  // --- layer 2: 64 -> 128 (WMMA), B direct from global (pre-swizzled) ---
  {
    const v16h* B2 = (const v16h*)w2s + lane;   // [frag][lane][16]
    v16h a2[2];
    #pragma unroll
    for (int kt = 0; kt < 2; ++kt) a2[kt] = load_A_frag(sH1, rlo, H1DIM, kt * 32, lane);

    for (int nt = 0; nt < H2DIM / 16; ++nt) {
      v8f c = {};
      #pragma unroll
      for (int kt = 0; kt < 2; ++kt)
        c = __builtin_amdgcn_wmma_f32_16x16x32_f16(false, a2[kt], false, B2[(nt*2+kt)*32],
                                                   (short)0, c, false, false);
      int col = nt * 16 + (lane & 15);
      float bb = b2[col];
      int rowAdd = 8 * (lane >> 4);
      #pragma unroll
      for (int r = 0; r < 8; ++r) {
        int row = rlo + r + rowAdd;             // C layout: VGPR r = M r / r+8 per half-wave
        sH2[row * H2DIM + col] = (_Float16)fmaxf(c[r] + bb, 0.0f);
      }
    }
  }
  __syncthreads();

  // --- layer 3: 128 -> 1024 (WMMA) + fused maxpool; no LDS in the loop ---
  {
    const v16h* B3 = (const v16h*)w3s + lane;   // [frag][lane][16]
    v16h a3[4];
    #pragma unroll
    for (int kt = 0; kt < 4; ++kt) a3[kt] = load_A_frag(sH2, rlo, H2DIM, kt * 32, lane);

    for (int nt = 0; nt < KFEAT / 16; ++nt) {
      if (nt + 1 < KFEAT / 16)
        __builtin_prefetch(w3s + (nt + 1) * 2048 + tid * 16, 0, 1);  // next 4KB frag group
      v8f c = {};
      #pragma unroll
      for (int kt = 0; kt < 4; ++kt)
        c = __builtin_amdgcn_wmma_f32_16x16x32_f16(false, a3[kt], false, B3[(nt*4+kt)*32],
                                                   (short)0, c, false, false);
      // maxpool over the tile's 16 rows: 8 VGPRs per lane, then across half-waves
      float m = c[0];
      #pragma unroll
      for (int r = 1; r < 8; ++r) m = fmaxf(m, c[r]);
      m = fmaxf(m, __shfl_xor(m, 16, 32));
      if (lane < 16) atomicMaxFloat(outbase + d.out_off + nt * 16 + lane, m);
    }
  }
}

__global__ void pinv_kernel(const float* __restrict__ f0, const float* __restrict__ fj,
                            const float* __restrict__ dt, float* __restrict__ pinv) {
  int b = blockIdx.x, tid = threadIdx.x;
  __shared__ float red[256];
  __shared__ float H[36], Hi[36];
  float idt[6];
  #pragma unroll
  for (int j = 0; j < 6; ++j) idt[j] = 1.0f / dt[j];

  float hloc[21];
  #pragma unroll
  for (int i = 0; i < 21; ++i) hloc[i] = 0.f;
  for (int n = tid; n < KFEAT; n += 256) {
    float f0v = f0[b * KFEAT + n];
    float jv[6];
    #pragma unroll
    for (int j = 0; j < 6; ++j) jv[j] = (f0v - fj[((b * 6 + j) << 10) + n]) * idt[j];
    int c = 0;
    #pragma unroll
    for (int i = 0; i < 6; ++i)
      #pragma unroll
      for (int j = i; j < 6; ++j) hloc[c++] += jv[i] * jv[j];
  }
  int c = 0;
  for (int i = 0; i < 6; ++i)
    for (int j = i; j < 6; ++j) {
      red[tid] = hloc[c];
      __syncthreads();
      for (int s = 128; s > 0; s >>= 1) { if (tid < s) red[tid] += red[tid + s]; __syncthreads(); }
      if (tid == 0) { H[i*6+j] = red[0]; H[j*6+i] = red[0]; }
      __syncthreads();
      ++c;
    }
  if (tid == 0) invert6(H, Hi);
  __syncthreads();
  for (int n = tid; n < KFEAT; n += 256) {
    float f0v = f0[b * KFEAT + n];
    float jv[6];
    #pragma unroll
    for (int j = 0; j < 6; ++j) jv[j] = (f0v - fj[((b * 6 + j) << 10) + n]) * idt[j];
    #pragma unroll
    for (int i = 0; i < 6; ++i) {
      float s = 0.f;
      #pragma unroll
      for (int j = 0; j < 6; ++j) s += Hi[i*6+j] * jv[j];
      pinv[(b * 6 + i) * KFEAT + n] = s;
    }
  }
}

__global__ void update_kernel(const float* __restrict__ f0, const float* __restrict__ f1,
                              const float* __restrict__ pinv, float* __restrict__ g,
                              float* __restrict__ out_r, int last) {
  int b = blockIdx.x, tid = threadIdx.x;
  __shared__ float rbuf[KFEAT];
  __shared__ float red[256];
  for (int n = tid; n < KFEAT; n += 256) {
    float rv = f1[b * KFEAT + n] - f0[b * KFEAT + n];
    rbuf[n] = rv;
    if (last) out_r[b * KFEAT + n] = rv;
  }
  __syncthreads();
  float dxs[6];
  for (int j = 0; j < 6; ++j) {
    float p = 0.f;
    for (int n = tid; n < KFEAT; n += 256) p += pinv[(b * 6 + j) * KFEAT + n] * rbuf[n];
    red[tid] = p;
    __syncthreads();
    for (int s = 128; s > 0; s >>= 1) { if (tid < s) red[tid] += red[tid + s]; __syncthreads(); }
    if (tid == 0) dxs[j] = -red[0];
    __syncthreads();
  }
  if (tid == 0) {
    float Re[9], pe[3];
    se3_exp_dev(dxs, Re, pe);
    float* gb = g + b * 12;
    float Rg[9], tg[3];
    #pragma unroll
    for (int i = 0; i < 9; ++i) Rg[i] = gb[i];
    tg[0] = gb[9]; tg[1] = gb[10]; tg[2] = gb[11];
    #pragma unroll
    for (int i = 0; i < 3; ++i) {
      #pragma unroll
      for (int j = 0; j < 3; ++j) {
        float s = 0.f;
        #pragma unroll
        for (int k = 0; k < 3; ++k) s += Re[i*3+k] * Rg[k*3+j];
        gb[i*3+j] = s;
      }
      gb[9+i] = Re[i*3]*tg[0] + Re[i*3+1]*tg[1] + Re[i*3+2]*tg[2] + pe[i];
    }
  }
}

__global__ void estg_kernel(const float* __restrict__ g, const float* __restrict__ means,
                            const float* __restrict__ igt_twist,
                            float* __restrict__ estg, float* __restrict__ invigt) {
  int b = threadIdx.x;
  if (b < 8) {
    const float* gb  = g + b * 12;
    const float* p0m = means + b * 3;
    const float* p1m = means + 24 + b * 3;
    float* e = estg + b * 12;
    #pragma unroll
    for (int i = 0; i < 9; ++i) e[i] = gb[i];
    #pragma unroll
    for (int i = 0; i < 3; ++i)
      e[9+i] = gb[9+i] + p0m[i] - (gb[i*3]*p1m[0] + gb[i*3+1]*p1m[1] + gb[i*3+2]*p1m[2]);
    float x[6];
    #pragma unroll
    for (int i = 0; i < 6; ++i) x[i] = igt_twist[b * 6 + i];
    float R[9], t[3];
    se3_exp_dev(x, R, t);
    float* v = invigt + b * 12;                 // inverse SE3: (R^T, -R^T t)
    #pragma unroll
    for (int i = 0; i < 3; ++i)
      #pragma unroll
      for (int j = 0; j < 3; ++j) v[i*3+j] = R[j*3+i];
    #pragma unroll
    for (int i = 0; i < 3; ++i)
      v[9+i] = -(R[0*3+i]*t[0] + R[1*3+i]*t[1] + R[2*3+i]*t[2]);
  }
}

__global__ void loss_kernel(const float* __restrict__ p_src, const float* __restrict__ estg,
                            const float* __restrict__ invigt, float* __restrict__ loss) {
  int idx = blockIdx.x * 256 + threadIdx.x;     // covers BATCH*NPTS exactly
  int b = idx >> 11, n = idx & 2047;
  const float* e = estg + b * 12;
  const float* v = invigt + b * 12;
  const float* p = p_src + (b * NPTS + n) * 3;
  float x = p[0], y = p[1], z = p[2];
  float s = 0.f;
  #pragma unroll
  for (int i = 0; i < 3; ++i) {
    float a = e[i*3]*x + e[i*3+1]*y + e[i*3+2]*z + e[9+i];
    float c = v[i*3]*x + v[i*3+1]*y + v[i*3+2]*z + v[9+i];
    s += fabsf(a - c);
  }
  __shared__ float red[256];
  red[threadIdx.x] = s;
  __syncthreads();
  for (int st = 128; st > 0; st >>= 1) {
    if (threadIdx.x < st) red[threadIdx.x] += red[threadIdx.x + st];
    __syncthreads();
  }
  if (threadIdx.x == 0) atomicAdd(loss, red[0] * (1.0f / (BATCH * NPTS * 3)));
}

// ---------------- launch ----------------
extern "C" void kernel_launch(void* const* d_in, const int* in_sizes, int n_in,
                              void* d_out, int out_size, void* d_ws, size_t ws_size,
                              hipStream_t stream) {
  const float* p_src = (const float*)d_in[0];
  const float* p_tgt = (const float*)d_in[1];
  const float* igt   = (const float*)d_in[2];
  const float* dt    = (const float*)d_in[3];
  const float* W1    = (const float*)d_in[4];
  const float* b1    = (const float*)d_in[5];
  const float* W2    = (const float*)d_in[6];
  const float* b2    = (const float*)d_in[7];
  const float* W3    = (const float*)d_in[8];
  // d_in[9] (b3) unused: it cancels in r, J, and the loss (max-pool commutes with
  // a per-column constant; only feature differences are ever consumed).

  float* out = (float*)d_out;                   // r: [0,8192), loss: [8192]
  float* ws  = (float*)d_ws;

  float* means  = ws + OFF_MEANS;
  float* transf = ws + OFF_TRANSF;
  float* g      = ws + OFF_G;
  float* featB  = ws + OFF_FEATB;               // f0 then fj, contiguous
  float* f1     = ws + OFF_F1;
  float* pinv   = ws + OFF_PINV;
  float* estg   = ws + OFF_ESTG;
  float* invigt = ws + OFF_INVIGT;
  CloudDesc* descB = (CloudDesc*)(ws + OFF_DESCB);
  CloudDesc* desc1 = (CloudDesc*)(ws + OFF_DESC1);
  _Float16* w2s    = (_Float16*)(ws + OFF_W2S);
  _Float16* w3s    = (_Float16*)(ws + OFF_W3S);

  wswz_kernel<<<544, 256, 0, stream>>>(W2, W3, w2s, w3s);   // (8192+131072)/256
  means_kernel<<<16, 256, 0, stream>>>(p_tgt, p_src, means);
  init_kernel<<<224, 256, 0, stream>>>(featB, out + BATCH * KFEAT);
  prepB_kernel<<<1, 64, 0, stream>>>(dt, means, transf, g, descB);

  // f0 + fj: 56 clouds x 32 row-tiles
  encode_kernel<<<56 * 32, ENC_THREADS, 0, stream>>>(p_tgt, descB, featB, W1, b1, b2, w2s, w3s);

  pinv_kernel<<<8, 256, 0, stream>>>(featB, featB + BATCH * KFEAT, dt, pinv);

  for (int it = 0; it < 5; ++it) {
    iter_setup_kernel<<<1, 256, 0, stream>>>(g, means, desc1, f1);
    encode_kernel<<<8 * 32, ENC_THREADS, 0, stream>>>(p_src, desc1, f1, W1, b1, b2, w2s, w3s);
    update_kernel<<<8, 256, 0, stream>>>(featB, f1, pinv, g, out, (it == 4) ? 1 : 0);
  }

  estg_kernel<<<1, 64, 0, stream>>>(g, means, igt, estg, invigt);
  loss_kernel<<<64, 256, 0, stream>>>(p_src, estg, invigt, out + BATCH * KFEAT);
}
